// TransConvLayer_20727512170562
// MI455X (gfx1250) — matrix operand order repaired and checked by reading, not compiled, not run on hardware
//
#include <hip/hip_runtime.h>
#include <hip/hip_bf16.h>
#include <math.h>

#define NROWS 100000
#define IN_CH 256
#define HD 512          // HEADS * DHEAD
#define HEADS 8
#define DHEAD 64
#define EPSV 1e-6f

typedef __attribute__((ext_vector_type(16))) __bf16 v16bf;
typedef __attribute__((ext_vector_type(8)))  float  v8f;

__device__ __forceinline__ v8f wmma_bf16(v16bf a, v16bf b, v8f c) {
  // 8 args: (neg_a, A, neg_b, B, c_mod, C, reuse_a, reuse_b)
  return __builtin_amdgcn_wmma_f32_16x16x32_bf16(false, a, false, b, (short)0, c,
                                                 false, false);
}

__device__ __forceinline__ v8f zero_v8f() {
  v8f z = {0.f, 0.f, 0.f, 0.f, 0.f, 0.f, 0.f, 0.f};
  return z;
}

// K offset for element pair i (0..7) of a 16x32 bf16 A fragment, given lane half.
// Lanes 0-15 hold K 0-15 (VGPR0-7); lanes 16-31 hold K 16-31.
__device__ __forceinline__ int a_koff(int i, int half) {
  return half * 16 + 2 * i;
}

// ---------------------------------------------------------------------------
// Kernel 0: zero stats region + convert the three weight matrices fp32 -> bf16
// ---------------------------------------------------------------------------
__global__ __launch_bounds__(256) void init_kernel(
    const float* __restrict__ wq, const float* __restrict__ wk,
    const float* __restrict__ wv, void* __restrict__ wb_v,
    float* __restrict__ stats) {
  __bf16* wb = (__bf16*)wb_v;
  int i = blockIdx.x * 256 + threadIdx.x;
  if (i < 131072)       wb[i] = (__bf16)wq[i];
  else if (i < 262144)  wb[i] = (__bf16)wk[i - 131072];
  else if (i < 393216)  wb[i] = (__bf16)wv[i - 262144];
  if (i < 33282) stats[i] = 0.f;   // [sumsq_q, sumsq_k, ks_sum(512), kvs(32768)]
}

// ---------------------------------------------------------------------------
// Kernel 1: C[N,512] = X[N,256] @ W[256,512] + b   (bf16 WMMA, f32 accumulate)
// 16 waves/block; each wave keeps its full-K B slice resident in VGPRs.
// A tile (16x256) staged fp32->bf16 through LDS, shared by all waves.
// Also accumulates sum(x^2) (q,k) and column sums (k) for the later passes.
// ---------------------------------------------------------------------------
__global__ __launch_bounds__(512) void qkv_gemm(
    const float* __restrict__ X, const void* __restrict__ wb_v,
    const float* __restrict__ bias, void* __restrict__ out_v,
    float* __restrict__ stats, int sumsq_idx, int do_colsum) {
  __shared__ __bf16 ldsA[16 * IN_CH];   // 8 KB
  const __bf16* wb = (const __bf16*)wb_v;
  __bf16* outb = (__bf16*)out_v;

  const int tid  = threadIdx.x;
  const int wave = tid >> 5;
  const int lane = tid & 31;
  const int m    = lane & 15;   // A row / B,C column within tile
  const int half = lane >> 4;
  const int ncol = blockIdx.y * 256 + wave * 16 + m;   // global output column

  // Resident B fragments (8 x 32x16 bf16 covering K=0..255 for this column set)
  v16bf bfr[8];
#pragma unroll
  for (int ks = 0; ks < 8; ++ks) {
    v16bf b;
#pragma unroll
    for (int i = 0; i < 8; ++i) {
      int kk = ks * 32 + half * 16 + 2 * i;
      b[2 * i]     = wb[(size_t)kk * HD + ncol];
      b[2 * i + 1] = wb[(size_t)(kk + 1) * HD + ncol];
    }
    bfr[ks] = b;
  }
  const float bv = bias[ncol];

  for (int rt = blockIdx.x; rt < NROWS / 16; rt += (int)gridDim.x) {
    const int row0 = rt * 16;
    {   // cooperative fp32 -> bf16 stage: 4096 elems, 512 threads, 8 each
      int base = tid * 8;
      int r = base >> 8, c = base & 255;
      const float* src = X + (size_t)(row0 + r) * IN_CH + c;
#pragma unroll
      for (int j = 0; j < 8; ++j) ldsA[r * IN_CH + c + j] = (__bf16)src[j];
      if (rt + (int)gridDim.x < NROWS / 16)   // speculative prefetch of next tile
        __builtin_prefetch(X + (size_t)(row0 + (int)gridDim.x * 16 + r) * IN_CH + c, 0, 1);
    }
    __syncthreads();

    v8f acc = zero_v8f();
#pragma unroll
    for (int ks = 0; ks < 8; ++ks) {
      v16bf a;
#pragma unroll
      for (int i = 0; i < 8; ++i) {
        int kk = ks * 32 + a_koff(i, half);
        a[2 * i]     = ldsA[m * IN_CH + kk];
        a[2 * i + 1] = ldsA[m * IN_CH + kk + 1];
      }
      acc = wmma_bf16(a, bfr[ks], acc);
    }

    const int rowbase = half * 8;
    float sq = 0.f, cs = 0.f;
#pragma unroll
    for (int r = 0; r < 8; ++r) {
      float v = acc[r] + bv;
      outb[(size_t)(row0 + rowbase + r) * HD + ncol] = (__bf16)v;
      sq += v * v;
      cs += v;
    }
    if (sumsq_idx >= 0) {   // global Frobenius-norm partial
#pragma unroll
      for (int off = 16; off > 0; off >>= 1) sq += __shfl_xor(sq, off, 32);
      if (lane == 0) atomicAdd(&stats[sumsq_idx], sq);
    }
    if (do_colsum) {        // ks_sum partial (lane L and L+16 share a column)
      cs += __shfl_xor(cs, 16, 32);
      if (lane < 16) atomicAdd(&stats[2 + ncol], cs);
    }
    __syncthreads();
  }
}

// ---------------------------------------------------------------------------
// Kernel 2: kvs[h,m,d] = sum_n k[n,h,m] * v[n,h,d]   (K^T @ V per head)
// Block = 4 waves = the 4 m-tiles of one head; 32x64 K and V chunks are staged
// into LDS with coalesced b128 loads (reused by all 4 waves), fragments are
// gathered from LDS. 4 WMMA accumulators per wave => 16x64 per wave.
// ---------------------------------------------------------------------------
__global__ __launch_bounds__(128) void kv_outer(
    const void* __restrict__ kb_v, const void* __restrict__ vb_v,
    float* __restrict__ kvs) {
  __shared__ __bf16 kT[32 * 64];   // chunk rows x head cols, 4 KB
  __shared__ __bf16 vT[32 * 64];   // 4 KB
  const __bf16* kb = (const __bf16*)kb_v;
  const __bf16* vb = (const __bf16*)vb_v;

  const int h    = blockIdx.x;
  const int tid  = threadIdx.x;
  const int mt   = tid >> 5;        // wave id = m-tile
  const int lane = tid & 31;
  const int m    = lane & 15;
  const int half = lane >> 4;
  const int hm   = h * DHEAD;
  const int CH   = NROWS / 125;     // 800 rows per block
  const int nb0  = blockIdx.y * CH;

  // staging map: thread t copies 16 consecutive bf16 of one (row, col) span
  const int sidx = tid * 16;
  const int srow = sidx >> 6, scol = sidx & 63;

  v8f acc[4];
#pragma unroll
  for (int dt = 0; dt < 4; ++dt) acc[dt] = zero_v8f();

  for (int nb = nb0; nb < nb0 + CH; nb += 32) {
    {   // coalesced global -> LDS stage of K and V chunks (32 rows x 64 cols)
      const __bf16* gk = kb + (size_t)(nb + srow) * HD + hm + scol;
      const __bf16* gv = vb + (size_t)(nb + srow) * HD + hm + scol;
#pragma unroll
      for (int j = 0; j < 16; ++j) {
        kT[sidx + j] = gk[j];
        vT[sidx + j] = gv[j];
      }
    }
    __syncthreads();

    v16bf a;   // A[m][kk] = K[nb+kk][hm + mt*16 + m]  (transposed gather, LDS)
#pragma unroll
    for (int i = 0; i < 8; ++i) {
      int kk = a_koff(i, half);
      a[2 * i]     = kT[kk * 64 + mt * 16 + m];
      a[2 * i + 1] = kT[(kk + 1) * 64 + mt * 16 + m];
    }
#pragma unroll
    for (int dt = 0; dt < 4; ++dt) {
      v16bf b;   // B[kk][d] = V[nb+kk][hm + dt*16 + d]  (from LDS)
#pragma unroll
      for (int i = 0; i < 8; ++i) {
        int kk = half * 16 + 2 * i;
        b[2 * i]     = vT[kk * 64 + dt * 16 + m];
        b[2 * i + 1] = vT[(kk + 1) * 64 + dt * 16 + m];
      }
      acc[dt] = wmma_bf16(a, b, acc[dt]);
    }
    __syncthreads();
  }

  const int rowbase = half * 8;
#pragma unroll
  for (int dt = 0; dt < 4; ++dt)
#pragma unroll
    for (int r = 0; r < 8; ++r)
      atomicAdd(&kvs[h * 4096 + (mt * 16 + rowbase + r) * 64 + dt * 16 + m],
                acc[dt][r]);
}

// ---------------------------------------------------------------------------
// Kernel 3: scale = 1/(||q||+eps) * 1/(||k||+eps); fold into kvs and ks_sum.
// kvs is written bf16 *transposed per head* ([h][d][m]) so the final pass can
// load B fragments as contiguous 32B runs (2 x b128 per fragment).
// ---------------------------------------------------------------------------
__global__ __launch_bounds__(256) void scale_fold(
    float* __restrict__ stats, float* __restrict__ kssum_s,
    void* __restrict__ kvb_v) {
  __shared__ float s_sc;
  __bf16* kvb = (__bf16*)kvb_v;
  if (threadIdx.x == 0) {
    float iq = 1.f / (sqrtf(stats[0]) + EPSV);
    float ik = 1.f / (sqrtf(stats[1]) + EPSV);
    s_sc = iq * ik;
  }
  __syncthreads();
  const float sc = s_sc;
  const float* kvs = stats + 514;
  for (int i = threadIdx.x; i < 32768; i += 256) {
    int hbase = i & ~4095;           // h * 4096
    int mrow  = (i >> 6) & 63;
    int d     = i & 63;
    kvb[hbase + d * 64 + mrow] = (__bf16)(kvs[i] * sc);   // transposed store
  }
  for (int i = threadIdx.x; i < 512; i += 256) kssum_s[i] = stats[2 + i] * sc;
}

// ---------------------------------------------------------------------------
// Kernel 4: out[n,d] = mean_h( (q@kvs_h + N*v) / (q . ks_sum_h + N + eps) )
// 8 waves per block (wave = head), head-mean via LDS ds_add_f32.
// ---------------------------------------------------------------------------
__global__ __launch_bounds__(256) void attn_final(
    const void* __restrict__ qb_v, const void* __restrict__ vb_v,
    const void* __restrict__ kvb_v, const float* __restrict__ kssum_s,
    float* __restrict__ out) {
  __shared__ float out_s[16 * 64];
  __shared__ float denom_s[HEADS][16];
  const __bf16* qb  = (const __bf16*)qb_v;
  const __bf16* vb  = (const __bf16*)vb_v;
  const __bf16* kvb = (const __bf16*)kvb_v;

  const int tid  = threadIdx.x;
  const int h    = tid >> 5;
  const int lane = tid & 31;
  const int m    = lane & 15;
  const int half = lane >> 4;
  const int hc   = h * DHEAD;
  const int row0 = blockIdx.x * 16;

  for (int i = tid; i < 1024; i += 256) out_s[i] = 0.f;

  v8f acc[4];
#pragma unroll
  for (int dt = 0; dt < 4; ++dt) acc[dt] = zero_v8f();

#pragma unroll
  for (int kc = 0; kc < 2; ++kc) {
    v16bf a;   // A[m][kk] = q[row0+m][hc + kc*32 + kk]
#pragma unroll
    for (int i = 0; i < 8; ++i) {
      int kk = kc * 32 + a_koff(i, half);
      a[2 * i]     = qb[(size_t)(row0 + m) * HD + hc + kk];
      a[2 * i + 1] = qb[(size_t)(row0 + m) * HD + hc + kk + 1];
    }
#pragma unroll
    for (int dt = 0; dt < 4; ++dt) {
      v16bf b;   // B[kk][d] = kvs_h[kk][dt*16+d]; kvb is [h][d][m] transposed,
                 // so per lane the 16 K-values are contiguous in memory.
#pragma unroll
      for (int i = 0; i < 8; ++i) {
        int kk = kc * 32 + half * 16 + 2 * i;
        b[2 * i]     = kvb[h * 4096 + (dt * 16 + m) * 64 + kk];
        b[2 * i + 1] = kvb[h * 4096 + (dt * 16 + m) * 64 + kk + 1];
      }
      acc[dt] = wmma_bf16(a, b, acc[dt]);
    }
  }

  {   // per-row normalizer: dot(q_row, ks_sum_scaled) + N + eps
    float dsum = 0.f;
    const __bf16* qrow = qb + (size_t)(row0 + m) * HD + hc + half * 32;
    const float* ks = kssum_s + hc + half * 32;
#pragma unroll
    for (int j = 0; j < 32; ++j) dsum += (float)qrow[j] * ks[j];
    dsum += __shfl_xor(dsum, 16, 32);
    if (lane < 16) denom_s[h][lane] = dsum + (float)NROWS + EPSV;
  }
  __syncthreads();

  const float Nf = (float)NROWS;
  const int rowbase = half * 8;
#pragma unroll
  for (int dt = 0; dt < 4; ++dt) {
#pragma unroll
    for (int r = 0; r < 8; ++r) {
      int row = rowbase + r;
      int d = dt * 16 + m;
      float vv  = (float)vb[(size_t)(row0 + row) * HD + hc + d];
      float num = acc[dt][r] + Nf * vv;
      atomicAdd(&out_s[row * 64 + d], 0.125f * num / denom_s[h][row]);
    }
  }
  __syncthreads();
  for (int i = tid; i < 1024; i += 256)
    out[(size_t)(row0 + (i >> 6)) * 64 + (i & 63)] = out_s[i];
}

// ---------------------------------------------------------------------------
extern "C" void kernel_launch(void* const* d_in, const int* in_sizes, int n_in,
                              void* d_out, int out_size, void* d_ws, size_t ws_size,
                              hipStream_t stream) {
  (void)in_sizes; (void)n_in; (void)out_size; (void)ws_size;
  const float* Xq = (const float*)d_in[0];
  const float* Xs = (const float*)d_in[1];
  const float* Wq = (const float*)d_in[2];
  const float* bq = (const float*)d_in[3];
  const float* Wk = (const float*)d_in[4];
  const float* bk = (const float*)d_in[5];
  const float* Wv = (const float*)d_in[6];
  const float* bv = (const float*)d_in[7];
  float* out = (float*)d_out;

  // Workspace layout (all offsets 256B-aligned), total ~308.2 MB:
  char* ws = (char*)d_ws;
  const size_t QKV_BYTES = (size_t)NROWS * HD * 2;   // 102,400,000 each
  void* qb = ws;
  void* kb = ws + QKV_BYTES;
  void* vb = ws + 2 * QKV_BYTES;
  char* wb = ws + 3 * QKV_BYTES;                     // 3 x [256x512] bf16
  float* stats  = (float*)(wb + (size_t)3 * 131072 * 2);   // 33282 f32
  float* kssum_s = (float*)((char*)stats + 133376);        // 512 f32 (scaled)
  void* kvb = (char*)kssum_s + 2048;                       // 32768 bf16 (scaled)

  init_kernel<<<1536, 256, 0, stream>>>(Wq, Wk, Wv, (void*)wb, stats);
  qkv_gemm<<<dim3(125, 2), 512, 0, stream>>>(Xq, wb,              bq, qb, stats, 0, 0);
  qkv_gemm<<<dim3(125, 2), 512, 0, stream>>>(Xs, wb + 131072 * 2, bk, kb, stats, 1, 1);
  qkv_gemm<<<dim3(125, 2), 512, 0, stream>>>(Xs, wb + 262144 * 2, bv, vb, stats, -1, 0);
  kv_outer<<<dim3(8, 125), 128, 0, stream>>>(kb, vb, stats + 514);
  scale_fold<<<1, 256, 0, stream>>>(stats, kssum_s, kvb);
  attn_final<<<6250, 256, 0, stream>>>(qb, vb, kvb, kssum_s, out);
}
